// GraphSAGE_35158602285296
// MI455X (gfx1250) — matrix operand order repaired and checked by reading, not compiled, not run on hardware
//
#include <hip/hip_runtime.h>
#include <math.h>
#include <stdint.h>

typedef __attribute__((ext_vector_type(16))) _Float16 v16h;
typedef __attribute__((ext_vector_type(8)))  float    v8f;

#define NNODES 40000
#define FEAT   128
#define HID    128
#define EMB    64
#define NEDGES 640000
#define KPAD   132   // 128 + 4 floats row padding to spread LDS banks

// ---- CDNA5 async global->LDS DMA helpers ------------------------------
// ISA 10.2: generic pointers to LDS carry the aperture tag in bits [63:32];
// the low 32 bits are the LDS byte offset the async VDST operand expects.
__device__ __forceinline__ uint32_t lds_off(const void* p) {
    return (uint32_t)(uintptr_t)p;
}

__device__ __forceinline__ void async_load_b128(void* lds_dst, const float* gsrc) {
    asm volatile("global_load_async_to_lds_b128 %0, %1, off"
                 :
                 : "v"(lds_off(lds_dst)), "v"(gsrc)
                 : "memory");
}

__device__ __forceinline__ void wait_asynccnt0() {
#if __has_builtin(__builtin_amdgcn_s_wait_asynccnt)
    __builtin_amdgcn_s_wait_asynccnt(0);
#else
    asm volatile("s_wait_asynccnt 0x0" ::: "memory");
#endif
}

// ---------------- degree / normalize ----------------
__global__ void degree_kernel(const int* __restrict__ dst, float* __restrict__ cnt, int E) {
    int e = blockIdx.x * blockDim.x + threadIdx.x;
    if (e < E) atomicAdd(&cnt[dst[e]], 1.0f);
}

__global__ void invcnt_kernel(float* __restrict__ cnt, int n) {
    int i = blockIdx.x * blockDim.x + threadIdx.x;
    if (i < n) cnt[i] = 1.0f / fmaxf(cnt[i], 1.0f);
}

// ---------------- edge scatter: agg[dst] += feat[src] ----------------
// One wave32 per edge; each lane moves 4 contiguous floats (float4 gather,
// 4x global_atomic_add_f32). Whole working set lives in the 192MB L2.
__global__ __launch_bounds__(256) void scatter_kernel(const float* __restrict__ feat,
                                                      const int* __restrict__ src,
                                                      const int* __restrict__ dst,
                                                      float* __restrict__ agg, int E) {
    int gid  = blockIdx.x * blockDim.x + threadIdx.x;
    int e    = gid >> 5;
    int lane = gid & 31;
    if (e >= E) return;
    int s = src[e];
    int d = dst[e];
    const float4 v = *(const float4*)(feat + (size_t)s * FEAT + lane * 4);
    float* o = agg + (size_t)d * FEAT + lane * 4;
    atomicAdd(o + 0, v.x);
    atomicAdd(o + 1, v.y);
    atomicAdd(o + 2, v.z);
    atomicAdd(o + 3, v.w);
}

// ---------------- fused dual-GEMM SAGE layer ----------------
// out[r, c] = sum_k scale1[r]*A1[r,k]*W1[c,k] + sum_k A2[r,k]*W2[c,k] + bias[c]
// One block = BANDS row bands x TPB column tiles (8 waves). A panels staged
// once per block into LDS via async DMA; one wave32 per 16x16 output tile,
// v_wmma_f32_16x16x32_f16, fragments via ds_load_b128.
template <int K, int TPB>
__global__ __launch_bounds__(256) void sage_gemm_kernel(
    const float* __restrict__ A1, const float* __restrict__ scale1,
    const float* __restrict__ W1,   // [O, K] row-major (== B^T)
    const float* __restrict__ A2,
    const float* __restrict__ W2,   // [O, K] row-major
    const float* __restrict__ bias,
    float* __restrict__ out,
    int elu)
{
    constexpr int BANDS = 8 / TPB;
    constexpr int O = TPB * 16;
    __shared__ float lds[BANDS][2][16][KPAD];

    const int tid  = threadIdx.x;
    const int lane = tid & 31;
    const int wave = tid >> 5;
    const int blockRow0 = blockIdx.x * (BANDS * 16);

    // ---- stage A1/A2 panels (BANDS x 2 x 16 x K floats) via async DMA ----
    constexpr int CW = K / 4;                       // 16B chunks per row
    constexpr int CHUNKS = BANDS * 2 * 16 * CW;
#pragma unroll
    for (int i = 0; i < CHUNKS / 256; ++i) {
        int c    = tid + i * 256;
        int band = c / (2 * 16 * CW);
        int rem  = c % (2 * 16 * CW);
        int mat  = rem / (16 * CW);
        int rc   = rem % (16 * CW);
        int r    = rc / CW;
        int cw   = rc % CW;
        const float* g = (mat == 0 ? A1 : A2) +
                         (size_t)(blockRow0 + band * 16 + r) * K + cw * 4;
        async_load_b128(&lds[band][mat][r][cw * 4], g);
    }
    wait_asynccnt0();
    __syncthreads();

    const int bw   = wave / TPB;            // row band of this wave
    const int ct   = wave % TPB;            // column tile of this wave
    const int col0 = ct << 4;
    const int row0 = blockRow0 + (bw << 4);

    const int n     = lane & 15;            // A row in tile / B+D column
    const int koff  = (lane < 16) ? 0 : 8;  // A-fragment K sub-offset
    const int kgrp  = (lane < 16) ? 0 : 8;  // B-fragment K sub-offset
    const int rbase = (lane < 16) ? 0 : 8;  // D-fragment row base

    const float s1 = scale1[row0 + n];
    const float* w1row = W1 + (size_t)(col0 + n) * K;
    const float* w2row = W2 + (size_t)(col0 + n) * K;

    v8f acc = {};
#pragma unroll
    for (int kb = 0; kb < K; kb += 32) {
        v16h a, b;
        // ---- A1 fragment from LDS (scaled): K runs [koff..+7],[16+koff..+7]
        {
            const float* p = &lds[bw][0][n][kb + koff];
            float4 f0 = *(const float4*)(p + 0);
            float4 f1 = *(const float4*)(p + 4);
            float4 f2 = *(const float4*)(p + 16);
            float4 f3 = *(const float4*)(p + 20);
            a[0]=(_Float16)(f0.x*s1); a[1]=(_Float16)(f0.y*s1); a[2]=(_Float16)(f0.z*s1); a[3]=(_Float16)(f0.w*s1);
            a[4]=(_Float16)(f1.x*s1); a[5]=(_Float16)(f1.y*s1); a[6]=(_Float16)(f1.z*s1); a[7]=(_Float16)(f1.w*s1);
            a[8]=(_Float16)(f2.x*s1); a[9]=(_Float16)(f2.y*s1); a[10]=(_Float16)(f2.z*s1); a[11]=(_Float16)(f2.w*s1);
            a[12]=(_Float16)(f3.x*s1); a[13]=(_Float16)(f3.y*s1); a[14]=(_Float16)(f3.z*s1); a[15]=(_Float16)(f3.w*s1);
        }
        // ---- B1 fragment from global (weights stay hot in WGP$/L2)
        {
            const float* p = w1row + kb + kgrp;
            float4 f0 = *(const float4*)(p + 0);
            float4 f1 = *(const float4*)(p + 4);
            float4 f2 = *(const float4*)(p + 16);
            float4 f3 = *(const float4*)(p + 20);
            b[0]=(_Float16)f0.x; b[1]=(_Float16)f0.y; b[2]=(_Float16)f0.z; b[3]=(_Float16)f0.w;
            b[4]=(_Float16)f1.x; b[5]=(_Float16)f1.y; b[6]=(_Float16)f1.z; b[7]=(_Float16)f1.w;
            b[8]=(_Float16)f2.x; b[9]=(_Float16)f2.y; b[10]=(_Float16)f2.z; b[11]=(_Float16)f2.w;
            b[12]=(_Float16)f3.x; b[13]=(_Float16)f3.y; b[14]=(_Float16)f3.z; b[15]=(_Float16)f3.w;
        }
        acc = __builtin_amdgcn_wmma_f32_16x16x32_f16(false, a, false, b,
                                                     (short)0, acc, false, false);

        // ---- A2 / W2 contribution (self term, unscaled)
        {
            const float* p = &lds[bw][1][n][kb + koff];
            float4 f0 = *(const float4*)(p + 0);
            float4 f1 = *(const float4*)(p + 4);
            float4 f2 = *(const float4*)(p + 16);
            float4 f3 = *(const float4*)(p + 20);
            a[0]=(_Float16)f0.x; a[1]=(_Float16)f0.y; a[2]=(_Float16)f0.z; a[3]=(_Float16)f0.w;
            a[4]=(_Float16)f1.x; a[5]=(_Float16)f1.y; a[6]=(_Float16)f1.z; a[7]=(_Float16)f1.w;
            a[8]=(_Float16)f2.x; a[9]=(_Float16)f2.y; a[10]=(_Float16)f2.z; a[11]=(_Float16)f2.w;
            a[12]=(_Float16)f3.x; a[13]=(_Float16)f3.y; a[14]=(_Float16)f3.z; a[15]=(_Float16)f3.w;
        }
        {
            const float* p = w2row + kb + kgrp;
            float4 f0 = *(const float4*)(p + 0);
            float4 f1 = *(const float4*)(p + 4);
            float4 f2 = *(const float4*)(p + 16);
            float4 f3 = *(const float4*)(p + 20);
            b[0]=(_Float16)f0.x; b[1]=(_Float16)f0.y; b[2]=(_Float16)f0.z; b[3]=(_Float16)f0.w;
            b[4]=(_Float16)f1.x; b[5]=(_Float16)f1.y; b[6]=(_Float16)f1.z; b[7]=(_Float16)f1.w;
            b[8]=(_Float16)f2.x; b[9]=(_Float16)f2.y; b[10]=(_Float16)f2.z; b[11]=(_Float16)f2.w;
            b[12]=(_Float16)f3.x; b[13]=(_Float16)f3.y; b[14]=(_Float16)f3.z; b[15]=(_Float16)f3.w;
        }
        acc = __builtin_amdgcn_wmma_f32_16x16x32_f16(false, a, false, b,
                                                     (short)0, acc, false, false);
    }

    // bias is per-column -> one scalar per lane; ELU optional; store D.
    const float bb = bias[col0 + n];
    float* op = out + (size_t)(row0 + rbase) * O + col0 + n;
#pragma unroll
    for (int v = 0; v < 8; ++v) {
        float r = acc[v] + bb;
        if (elu) r = (r > 0.0f) ? r : (__expf(r) - 1.0f);
        op[(size_t)v * O] = r;
    }
}

// ---------------- row-wise log_softmax over 64 columns ----------------
__global__ __launch_bounds__(256) void logsoftmax_kernel(float* __restrict__ out, int Nrows) {
    int row  = blockIdx.x * (blockDim.x >> 5) + (threadIdx.x >> 5);
    int lane = threadIdx.x & 31;
    if (row >= Nrows) return;
    float* p = out + (size_t)row * EMB;
    float v0 = p[lane], v1 = p[lane + 32];
    float m = fmaxf(v0, v1);
#pragma unroll
    for (int off = 16; off > 0; off >>= 1) m = fmaxf(m, __shfl_xor(m, off, 32));
    float s = __expf(v0 - m) + __expf(v1 - m);
#pragma unroll
    for (int off = 16; off > 0; off >>= 1) s += __shfl_xor(s, off, 32);
    float l = m + __logf(s);
    p[lane]      = v0 - l;
    p[lane + 32] = v1 - l;
}

extern "C" void kernel_launch(void* const* d_in, const int* in_sizes, int n_in,
                              void* d_out, int out_size, void* d_ws, size_t ws_size,
                              hipStream_t stream) {
    const float* x    = (const float*)d_in[0];
    const int*   edge = (const int*)d_in[1];      // [2, E] int32
    const float* W1_l = (const float*)d_in[2];    // [128,128]
    const float* b1   = (const float*)d_in[3];    // [128]
    const float* W1_r = (const float*)d_in[4];    // [128,128]
    const float* W2_l = (const float*)d_in[5];    // [64,128]
    const float* b2   = (const float*)d_in[6];    // [64]
    const float* W2_r = (const float*)d_in[7];    // [64,128]
    const int* src = edge;
    const int* dst = edge + NEDGES;

    // workspace carve (floats, 16B-aligned offsets)
    float* ws  = (float*)d_ws;
    float* cnt = ws;                               // 40000 -> pad to 40960
    float* agg = ws + 40960;                       // 40000*128 (layers 1 and 2)
    float* h   = agg + (size_t)NNODES * FEAT;      // 40000*128

    const size_t featBytes = (size_t)NNODES * FEAT * sizeof(float);

    // degree -> 1/max(deg,1)
    hipMemsetAsync(cnt, 0, NNODES * sizeof(float), stream);
    hipMemsetAsync(agg, 0, featBytes, stream);
    degree_kernel<<<(NEDGES + 255) / 256, 256, 0, stream>>>(dst, cnt, NEDGES);
    invcnt_kernel<<<(NNODES + 255) / 256, 256, 0, stream>>>(cnt, NNODES);

    // layer 1: scatter x, then h = elu(mean_agg @ W1_l^T + b1 + x @ W1_r^T)
    scatter_kernel<<<(NEDGES * 32) / 256, 256, 0, stream>>>(x, src, dst, agg, NEDGES);
    sage_gemm_kernel<FEAT, HID / 16><<<NNODES / 16, 256, 0, stream>>>(
        agg, cnt, W1_l, x, W1_r, b1, h, /*elu=*/1);

    // layer 2: scatter h, then out = mean_agg @ W2_l^T + b2 + h @ W2_r^T
    hipMemsetAsync(agg, 0, featBytes, stream);
    scatter_kernel<<<(NEDGES * 32) / 256, 256, 0, stream>>>(h, src, dst, agg, NEDGES);
    sage_gemm_kernel<HID, EMB / 16><<<NNODES / 32, 256, 0, stream>>>(
        agg, cnt, W2_l, h, W2_r, b2, (float*)d_out, /*elu=*/0);

    // log_softmax rows of 64, one wave per row
    logsoftmax_kernel<<<(NNODES + 7) / 8, 256, 0, stream>>>((float*)d_out, NNODES);
}
